// AttnDecoder_88923002896698
// MI455X (gfx1250) — matrix operand order repaired and checked by reading, not compiled, not run on hardware
//
#include <hip/hip_runtime.h>
#include <hip/hip_bf16.h>

// Problem constants
#define B_  32
#define S_  128
#define T_  64
#define E_  512
#define H_  512
#define IN_ 1024
#define V_  32000
#define G_  2048   // 4*H

typedef unsigned short u16;
typedef __attribute__((ext_vector_type(16))) __bf16 v16bf;
typedef __attribute__((ext_vector_type(8)))  __bf16 v8bf;
typedef __attribute__((ext_vector_type(8)))  float  v8f;
typedef __attribute__((ext_vector_type(4)))  unsigned int u32x4;
typedef __attribute__((ext_vector_type(8)))  int i32x8;
typedef __attribute__((ext_vector_type(4)))  int i32x4;

#if defined(__HIP_DEVICE_COMPILE__) && __has_builtin(__builtin_amdgcn_tensor_load_to_lds) && \
    __has_builtin(__builtin_amdgcn_s_wait_tensorcnt)
#define HAS_TDM 1
#else
#define HAS_TDM 0
#endif

__device__ __forceinline__ u16 f2bf(float f) {
  unsigned u = __float_as_uint(f);
  unsigned r = (u + 0x7FFFu + ((u >> 16) & 1u)) >> 16;   // RNE
  return (u16)r;
}
__device__ __forceinline__ float sigm(float x) { return 1.0f / (1.0f + expf(-x)); }

// CDNA5 wave32 WMMA bf16 fragment loaders (per ISA 7.12.2 layouts).
// A (16x32, MxK): lane -> M = lane&15; half = lane>>4; elements 0..7 hold
// K = kt+8*half .. +7, elements 8..15 hold K = kt+16+8*half .. +7.
__device__ __forceinline__ v16bf load_a_frag(const u16* __restrict__ row, int kt, int half) {
  v8bf a0 = *(const v8bf*)(row + kt + 8 * half);
  v8bf a1 = *(const v8bf*)(row + kt + 16 + 8 * half);
  return __builtin_shufflevector(a0, a1, 0,1,2,3,4,5,6,7,8,9,10,11,12,13,14,15);
}
// B (32x16, KxN): lane -> N = lane&15; elements 0..7 hold K = kt+8*half .. +7,
// elements 8..15 hold K = kt+16+8*half .. +7 (same as A).
__device__ __forceinline__ v16bf load_b_frag(const u16* __restrict__ row, int kt, int half) {
  v8bf b0 = *(const v8bf*)(row + kt + 8 * half);
  v8bf b1 = *(const v8bf*)(row + kt + 16 + 8 * half);
  return __builtin_shufflevector(b0, b1, 0,1,2,3,4,5,6,7,8,9,10,11,12,13,14,15);
}
#define WMMA_BF16(a, b, c) \
  __builtin_amdgcn_wmma_f32_16x16x32_bf16(false, (a), false, (b), (short)0, (c), false, false)

// ---------------------------------------------------------------------------
// K0: v_enc[j] = sum_k attn2_w[k] * attn1_w[k, j],  j < IN_  (enc part only;
// the h-dependent logit term is constant over s -> dropped before softmax)
__global__ void venc_kernel(const float* __restrict__ attn1_w,
                            const float* __restrict__ attn2_w,
                            float* __restrict__ venc) {
  int j = blockIdx.x * 256 + threadIdx.x;   // 0..1023
  float s = 0.f;
  for (int k = 0; k < G_; ++k) s += attn2_w[k] * attn1_w[k * G_ + j];
  venc[j] = s;
}

// K1: pre_enc[b,s] = enc_output[b,s,:] . v_enc
__global__ void preenc_kernel(const float* __restrict__ enc,
                              const float* __restrict__ venc,
                              float* __restrict__ pre) {
  __shared__ float red[128];
  int bs = blockIdx.x;                      // 0..4095
  const float* row = enc + (size_t)bs * IN_;
  float s = 0.f;
  for (int k = threadIdx.x; k < IN_; k += 128) s += row[k] * venc[k];
  red[threadIdx.x] = s;
  __syncthreads();
  for (int st = 64; st > 0; st >>= 1) {
    if (threadIdx.x < st) red[threadIdx.x] += red[threadIdx.x + st];
    __syncthreads();
  }
  if (threadIdx.x == 0) pre[bs] = red[0];
}

// K2: softmax over s + context (constant over time) -> bf16
__global__ void ctx_kernel(const float* __restrict__ pre,
                           const float* __restrict__ enc,
                           u16* __restrict__ ctxb) {
  __shared__ float w[S_];
  int b = blockIdx.x, tid = threadIdx.x;
  if (tid < S_) w[tid] = pre[b * S_ + tid];
  __syncthreads();
  float mx = -1e30f;
  for (int s = 0; s < S_; ++s) mx = fmaxf(mx, w[s]);
  __syncthreads();
  if (tid < S_) w[tid] = expf(w[tid] - mx);
  __syncthreads();
  float sum = 0.f;
  for (int s = 0; s < S_; ++s) sum += w[s];
  float inv = 1.f / sum;
  for (int j = tid; j < IN_; j += 256) {
    float acc = 0.f;
    for (int s = 0; s < S_; ++s) acc += w[s] * enc[((size_t)b * S_ + s) * IN_ + j];
    ctxb[b * IN_ + j] = f2bf(acc * inv);
  }
}

// K3: gather embeddings -> bf16, row m = t*32 + b
__global__ void gather_emb_kernel(const int* __restrict__ tokens,
                                  const float* __restrict__ emb,
                                  u16* __restrict__ dst) {
  int idx = blockIdx.x * 256 + threadIdx.x;   // < 2048*512
  int m = idx >> 9, k = idx & 511;
  int b = m & 31, t = m >> 5;
  int tok = tokens[b * T_ + t];
  dst[idx] = f2bf(emb[(size_t)tok * E_ + k]);
}

// K4a: gate-permuted bf16 weight convert.  q = 4u + gate  <-  orig = gate*512 + u
__global__ void cvt_perm_kernel(u16* __restrict__ dst, const float* __restrict__ src,
                                int K, int srcStride, int srcOff) {
  int idx = blockIdx.x * 256 + threadIdx.x;   // < 2048*K
  int q = idx / K, k = idx - q * K;
  int orig = (q & 3) * 512 + (q >> 2);
  dst[idx] = f2bf(src[(size_t)orig * srcStride + srcOff + k]);
}

// K4b: wcat1 = [w_ih1 | w_hh1] gate-permuted bf16, K = 1024
__global__ void cvt_wcat1_kernel(u16* __restrict__ dst,
                                 const float* __restrict__ w_ih1,
                                 const float* __restrict__ w_hh1) {
  int idx = blockIdx.x * 256 + threadIdx.x;   // < 2048*1024
  int q = idx >> 10, k = idx & 1023;
  int orig = (q & 3) * 512 + (q >> 2);
  float v = (k < 512) ? w_ih1[(size_t)orig * 512 + k] : w_hh1[(size_t)orig * 512 + (k - 512)];
  dst[idx] = f2bf(v);
}

// K4c: permuted layer-1 bias
__global__ void cvt_bias1_kernel(float* __restrict__ dst,
                                 const float* __restrict__ b_ih1,
                                 const float* __restrict__ b_hh1) {
  int q = blockIdx.x * 256 + threadIdx.x;     // < 2048
  int orig = (q & 3) * 512 + (q >> 2);
  dst[q] = b_ih1[orig] + b_hh1[orig];
}

// K4d: out_w f32 -> bf16 (vectorized)
__global__ void cvt_outw_kernel(u16* __restrict__ dst, const float* __restrict__ src) {
  int i4 = blockIdx.x * 256 + threadIdx.x;    // < 32000*512/4
  float4 v = ((const float4*)src)[i4];
  ushort4 o;
  o.x = f2bf(v.x); o.y = f2bf(v.y); o.z = f2bf(v.z); o.w = f2bf(v.w);
  ((ushort4*)dst)[i4] = o;
}

// K5: initial h/c state.  hbufA: [32 x 1024] bf16 (h0 cols 0..511, h1 cols 512..1023)
__global__ void init_state_kernel(const float* __restrict__ enc_h,
                                  const float* __restrict__ enc_c,
                                  u16* __restrict__ hbufA,
                                  float* __restrict__ c0, float* __restrict__ c1) {
  int idx = blockIdx.x * 256 + threadIdx.x;   // < 32*512
  int b = idx >> 9, j = idx & 511;
  hbufA[b * 1024 + j]       = f2bf(enc_h[b * H_ + j]);
  hbufA[b * 1024 + 512 + j] = f2bf(enc_h[B_ * H_ + b * H_ + j]);
  c0[idx] = enc_c[b * H_ + j];
  c1[idx] = enc_c[B_ * H_ + b * H_ + j];
}

// ---------------------------------------------------------------------------
// K6: gate pre-activations for all timesteps (gate-permuted layout):
// gp[m=t*32+b, q] = emb_m . wih0e[q] + ctx_b . wih0c[q] + (b_ih0+b_hh0)[orig(q)]
__global__ void gates_pre_gemm(const u16* __restrict__ embseq,
                               const u16* __restrict__ ctxb,
                               const u16* __restrict__ wih0e,
                               const u16* __restrict__ wih0c,
                               const float* __restrict__ b_ih0,
                               const float* __restrict__ b_hh0,
                               float* __restrict__ gp) {
  int lane = threadIdx.x & 31, wv = threadIdx.x >> 5;
  int n_tile = blockIdx.x * 8 + wv;           // 0..127
  int m_tile = blockIdx.y;                    // 0..127
  int nl = lane & 15, half = lane >> 4;
  int n = n_tile * 16 + nl;
  int orig = (n & 3) * 512 + (n >> 2);
  float bias = b_ih0[orig] + b_hh0[orig];
  v8f acc;
#pragma unroll
  for (int r = 0; r < 8; ++r) acc[r] = bias;

  const u16* arow1 = embseq + (size_t)(m_tile * 16 + nl) * E_;
  const u16* brow1 = wih0e + (size_t)n * E_;
#pragma unroll 4
  for (int kt = 0; kt < E_; kt += 32)
    acc = WMMA_BF16(load_a_frag(arow1, kt, half), load_b_frag(brow1, kt, half), acc);

  int bidx = (m_tile * 16 + nl) & 31;
  const u16* arow2 = ctxb + (size_t)bidx * IN_;
  const u16* brow2 = wih0c + (size_t)n * IN_;
#pragma unroll 4
  for (int kt = 0; kt < IN_; kt += 32)
    acc = WMMA_BF16(load_a_frag(arow2, kt, half), load_b_frag(brow2, kt, half), acc);

#pragma unroll
  for (int r = 0; r < 8; ++r) {
    int m = m_tile * 16 + 8 * half + r;
    gp[(size_t)m * G_ + n] = acc[r];
  }
}

// ---------------------------------------------------------------------------
// K7: one recurrent step, layer 0: gates = gp[t] + h0 @ Whh0^T, fused LSTM cell.
// Gate-interleaved layout: tile column c -> unit u = n_tile*4 + c/4, gate = c%4.
__global__ void step_l0(int t,
                        const u16* __restrict__ hread,    // h0 in cols 0..511
                        u16* __restrict__ hwrite,         // new h0 -> cols 0..511
                        const u16* __restrict__ whh0,
                        const float* __restrict__ gp,
                        float* __restrict__ c0) {
  __shared__ float tile[8][16][16];
  int lane = threadIdx.x & 31, wv = threadIdx.x >> 5;
  int wg = blockIdx.x * 8 + wv;               // 0..255
  int n_tile = wg & 127, m_tile = wg >> 7;    // 128 x 2
  int nl = lane & 15, half = lane >> 4;
  int n = n_tile * 16 + nl;

  const float* gprow = gp + (size_t)(t * B_) * G_;
  v8f acc;
#pragma unroll
  for (int r = 0; r < 8; ++r)
    acc[r] = gprow[(size_t)(m_tile * 16 + 8 * half + r) * G_ + n];

  const u16* arow = hread + (size_t)(m_tile * 16 + nl) * 1024;   // h0 region
  const u16* brow = whh0 + (size_t)n * H_;
#pragma unroll 4
  for (int kt = 0; kt < H_; kt += 32)
    acc = WMMA_BF16(load_a_frag(arow, kt, half), load_b_frag(brow, kt, half), acc);

#pragma unroll
  for (int r = 0; r < 8; ++r) tile[wv][8 * half + r][nl] = acc[r];
  __syncthreads();

  for (int o = lane; o < 64; o += 32) {
    int row = o & 15, uu = o >> 4;            // 16 rows x 4 units
    float gi = tile[wv][row][uu * 4 + 0];
    float gf = tile[wv][row][uu * 4 + 1];
    float gc = tile[wv][row][uu * 4 + 2];
    float go = tile[wv][row][uu * 4 + 3];
    int b = m_tile * 16 + row;
    int u = n_tile * 4 + uu;
    float cold = c0[b * H_ + u];
    float cn = sigm(gf) * cold + sigm(gi) * tanhf(gc);
    float h = sigm(go) * tanhf(cn);
    c0[b * H_ + u] = cn;
    hwrite[b * 1024 + u] = f2bf(h);
  }
}

// K8: layer 1: gates = [h0_new | h1_old] @ wcat1^T + bias1p, fused cell, emit dec.
__global__ void step_l1(int t,
                        const u16* __restrict__ h0buf,    // h0_new in cols 0..511
                        const u16* __restrict__ h1read,   // h1_old in cols 512..1023
                        u16* __restrict__ h1write,        // new h1 -> cols 512..1023
                        const u16* __restrict__ wcat1,
                        const float* __restrict__ bias1p,
                        float* __restrict__ c1,
                        float* __restrict__ dec,
                        u16* __restrict__ decb) {
  __shared__ float tile[8][16][16];
  int lane = threadIdx.x & 31, wv = threadIdx.x >> 5;
  int wg = blockIdx.x * 8 + wv;
  int n_tile = wg & 127, m_tile = wg >> 7;
  int nl = lane & 15, half = lane >> 4;
  int n = n_tile * 16 + nl;

  float bias = bias1p[n];
  v8f acc;
#pragma unroll
  for (int r = 0; r < 8; ++r) acc[r] = bias;

  const u16* arow0 = h0buf + (size_t)(m_tile * 16 + nl) * 1024;
  const u16* arow1 = h1read + (size_t)(m_tile * 16 + nl) * 1024;
  const u16* brow = wcat1 + (size_t)n * 1024;
#pragma unroll 4
  for (int kt = 0; kt < 1024; kt += 32) {
    const u16* arow = (kt < 512) ? arow0 : arow1;   // h1 stored at cols 512+
    acc = WMMA_BF16(load_a_frag(arow, kt, half), load_b_frag(brow, kt, half), acc);
  }

#pragma unroll
  for (int r = 0; r < 8; ++r) tile[wv][8 * half + r][nl] = acc[r];
  __syncthreads();

  for (int o = lane; o < 64; o += 32) {
    int row = o & 15, uu = o >> 4;
    float gi = tile[wv][row][uu * 4 + 0];
    float gf = tile[wv][row][uu * 4 + 1];
    float gc = tile[wv][row][uu * 4 + 2];
    float go = tile[wv][row][uu * 4 + 3];
    int b = m_tile * 16 + row;
    int u = n_tile * 4 + uu;
    float cold = c1[b * H_ + u];
    float cn = sigm(gf) * cold + sigm(gi) * tanhf(gc);
    float h = sigm(go) * tanhf(cn);
    c1[b * H_ + u] = cn;
    h1write[b * 1024 + 512 + u] = f2bf(h);
    size_t drow = (size_t)(b * T_ + t) * H_ + u;    // dec layout [B, T, H]
    dec[drow] = h;
    decb[drow] = f2bf(h);
  }
}

// ---------------------------------------------------------------------------
// K9: output projection: out[m, n] = dec_bf16[m,:] . out_w_bf16[n,:] + out_b[n]
// M = 2048 (m = b*64+t), N = 32000, K = 512.  4 N-tiles per wave (A reuse).
// The 16x512 bf16 A-tile (16 KB) is shared by all 4 waves of the block: it is
// staged into LDS by the Tensor Data Mover (tensor_load_to_lds, D# per ISA
// ch.8), tracked with TENSORcnt; fallback = cooperative global->LDS copy.
__global__ void final_gemm(const u16* __restrict__ decb,
                           const u16* __restrict__ outwb,
                           const float* __restrict__ out_b,
                           float* __restrict__ out) {
  __shared__ __align__(32) u16 atile[16 * H_];            // 16 KB
  int lane = threadIdx.x & 31, wv = threadIdx.x >> 5;     // 4 waves/block
  int m_tile = blockIdx.y;                                // 0..127
  int n_group = blockIdx.x * 4 + wv;                      // 0..499
  int n0 = n_group * 4;                                   // first of 4 n-tiles
  int nl = lane & 15, half = lane >> 4;

#if HAS_TDM
  if (threadIdx.x < 32) {
    // D# group 0: count=1 | lds_addr | global_addr(57b) | type=2
    unsigned lds_off = (unsigned)(size_t)&atile[0];       // generic addr low 32b = LDS offset
    unsigned long long ga =
        (unsigned long long)(const void*)(decb + (size_t)(m_tile * 16) * H_);
    u32x4 g0;
    g0[0] = 1u;                                           // count = 1 valid descriptor
    g0[1] = lds_off;
    g0[2] = (unsigned)(ga & 0xFFFFFFFFu);
    g0[3] = (unsigned)((ga >> 32) & 0x1FFFFFFu) | (2u << 30);   // type=2 (image)
    // D# group 1: data_size=2B; tensor 512 x 2048; tile 512 x 16; dim0_stride=512
    i32x8 g1;
    g1[0] = (int)(1u << 16);                              // data_size code 1 -> 2 bytes
    g1[1] = (int)(512u << 16);                            // tensor_dim0[15:0] in [63:48]
    g1[2] = (int)(2048u << 16);                           // tensor_dim1[15:0] in [95:80]
    g1[3] = (int)(512u << 16);                            // tile_dim0 in [127:112]
    g1[4] = 16;                                           // tile_dim1 in [143:128]
    g1[5] = 512;                                          // tensor_dim0_stride[31:0]
    g1[6] = 0;
    g1[7] = 0;
    i32x4 g2 = {0, 0, 0, 0}, g3 = {0, 0, 0, 0};           // dims 2..4 unused (2D)
    i32x8 g4 = {0, 0, 0, 0, 0, 0, 0, 0};                  // VADDR4 slot: unused -> NULL
    __builtin_amdgcn_tensor_load_to_lds(g0, g1, g2, g3, g4, 0);
    __builtin_amdgcn_s_wait_tensorcnt(0);
  }
#else
  {
    const uint4* src = (const uint4*)(decb + (size_t)(m_tile * 16) * H_);
    uint4* dst = (uint4*)&atile[0];
    for (int i = threadIdx.x; i < 16 * H_ * 2 / 16; i += 128) dst[i] = src[i];
  }
#endif
  __syncthreads();

  v8f acc[4];
#pragma unroll
  for (int j = 0; j < 4; ++j) {
    float bv = out_b[(n0 + j) * 16 + nl];
#pragma unroll
    for (int r = 0; r < 8; ++r) acc[j][r] = bv;
  }
  const u16* brows[4];
#pragma unroll
  for (int j = 0; j < 4; ++j) brows[j] = outwb + (size_t)((n0 + j) * 16 + nl) * H_;

#pragma unroll 2
  for (int kt = 0; kt < H_; kt += 32) {
    // A fragment from LDS (ds_load_b128 path)
    v8bf a0 = *(const v8bf*)&atile[nl * H_ + kt + 8 * half];
    v8bf a1 = *(const v8bf*)&atile[nl * H_ + kt + 16 + 8 * half];
    v16bf a = __builtin_shufflevector(a0, a1, 0,1,2,3,4,5,6,7,8,9,10,11,12,13,14,15);
    __builtin_prefetch(brows[0] + kt + 128, 0, 1);        // global_prefetch_b8
#pragma unroll
    for (int j = 0; j < 4; ++j) {
      v16bf b = load_b_frag(brows[j], kt, half);
      acc[j] = WMMA_BF16(a, b, acc[j]);
    }
  }

#pragma unroll
  for (int j = 0; j < 4; ++j) {
#pragma unroll
    for (int r = 0; r < 8; ++r) {
      int m = m_tile * 16 + 8 * half + r;
      int n = (n0 + j) * 16 + nl;
      out[(size_t)m * V_ + n] = acc[j][r];
    }
  }
}

// ---------------------------------------------------------------------------
extern "C" void kernel_launch(void* const* d_in, const int* in_sizes, int n_in,
                              void* d_out, int out_size, void* d_ws, size_t ws_size,
                              hipStream_t stream) {
  const int*   tokens  = (const int*)  d_in[0];
  const float* enc_out = (const float*)d_in[1];
  const float* enc_h   = (const float*)d_in[2];
  const float* enc_c   = (const float*)d_in[3];
  const float* emb     = (const float*)d_in[4];
  const float* attn1_w = (const float*)d_in[5];
  // d_in[6] attn1_b, d_in[8] attn2_b: shift-invariant under softmax -> unused
  const float* attn2_w = (const float*)d_in[7];
  const float* w_ih0   = (const float*)d_in[9];
  const float* w_hh0   = (const float*)d_in[10];
  const float* b_ih0   = (const float*)d_in[11];
  const float* b_hh0   = (const float*)d_in[12];
  const float* w_ih1   = (const float*)d_in[13];
  const float* w_hh1   = (const float*)d_in[14];
  const float* b_ih1   = (const float*)d_in[15];
  const float* b_hh1   = (const float*)d_in[16];
  const float* out_w   = (const float*)d_in[17];
  const float* out_b   = (const float*)d_in[18];
  float* out = (float*)d_out;

  // Workspace carve-up (~70 MB total)
  char* wp = (char*)d_ws;
  auto alloc = [&](size_t bytes) -> void* {
    void* p = (void*)wp;
    wp += (bytes + 255) & ~(size_t)255;
    return p;
  };
  float* venc    = (float*)alloc(IN_ * 4);
  float* pre     = (float*)alloc(B_ * S_ * 4);
  u16*   ctxb    = (u16*)  alloc(B_ * IN_ * 2);
  u16*   hbufA   = (u16*)  alloc(B_ * 1024 * 2);
  u16*   hbufB   = (u16*)  alloc(B_ * 1024 * 2);
  float* c0      = (float*)alloc(B_ * H_ * 4);
  float* c1      = (float*)alloc(B_ * H_ * 4);
  float* bias1p  = (float*)alloc(G_ * 4);
  u16*   whh0b   = (u16*)  alloc((size_t)G_ * H_ * 2);      // 2 MB
  u16*   wih0eb  = (u16*)  alloc((size_t)G_ * E_ * 2);      // 2 MB
  u16*   wih0cb  = (u16*)  alloc((size_t)G_ * IN_ * 2);     // 4 MB
  u16*   wcat1b  = (u16*)  alloc((size_t)G_ * 1024 * 2);    // 4 MB
  u16*   outwb   = (u16*)  alloc((size_t)V_ * H_ * 2);      // 32.8 MB
  u16*   embseqb = (u16*)  alloc((size_t)T_ * B_ * E_ * 2); // 2 MB
  u16*   decb    = (u16*)  alloc((size_t)B_ * T_ * H_ * 2); // 2 MB
  float* dec     = (float*)alloc((size_t)B_ * T_ * H_ * 4); // 4 MB
  float* gp      = (float*)alloc((size_t)T_ * B_ * G_ * 4); // 16.8 MB

  // --- Stage A: precompute (all parallel) ---
  venc_kernel      <<<4,     256, 0, stream>>>(attn1_w, attn2_w, venc);
  preenc_kernel    <<<B_*S_, 128, 0, stream>>>(enc_out, venc, pre);
  ctx_kernel       <<<B_,    256, 0, stream>>>(pre, enc_out, ctxb);
  gather_emb_kernel<<<4096,  256, 0, stream>>>(tokens, emb, embseqb);
  cvt_perm_kernel  <<<4096,  256, 0, stream>>>(whh0b,  w_hh0, 512, 512, 0);
  cvt_perm_kernel  <<<4096,  256, 0, stream>>>(wih0eb, w_ih0, 512, 1536, 0);
  cvt_perm_kernel  <<<8192,  256, 0, stream>>>(wih0cb, w_ih0, 1024, 1536, 512);
  cvt_wcat1_kernel <<<8192,  256, 0, stream>>>(wcat1b, w_ih1, w_hh1);
  cvt_bias1_kernel <<<8,     256, 0, stream>>>(bias1p, b_ih1, b_hh1);
  cvt_outw_kernel  <<<16000, 256, 0, stream>>>(outwb, out_w);
  init_state_kernel<<<64,    256, 0, stream>>>(enc_h, enc_c, hbufA, c0, c1);

  gates_pre_gemm<<<dim3(16, 128), 256, 0, stream>>>(embseqb, ctxb, wih0eb, wih0cb,
                                                    b_ih0, b_hh0, gp);

  // --- Stage B: 64-step recurrence, 2 fused GEMM+cell kernels per step ---
  u16* bufs[2] = {hbufA, hbufB};
  for (int t = 0; t < T_; ++t) {
    u16* rbuf = bufs[t & 1];
    u16* wbuf = bufs[(t + 1) & 1];
    step_l0<<<32, 256, 0, stream>>>(t, rbuf, wbuf, whh0b, gp, c0);
    step_l1<<<32, 256, 0, stream>>>(t, wbuf, rbuf, wbuf, wcat1b, bias1p, c1, dec, decb);
  }

  // --- Stage C: output projection [2048,512] x [512,32000] ---
  final_gemm<<<dim3(125, 128), 128, 0, stream>>>(decb, outwb, out_b, out);
}
